// HyenaOperator_22539988369649
// MI455X (gfx1250) — compile-verified, matrix operand order checked by reading, not run on hardware
//
#include <hip/hip_runtime.h>
#include <hip/hip_bf16.h>

// Problem constants (from reference)
#define B_     16
#define L_     4096
#define D_     256
#define INNER_ 768
#define FORD_  64
#define NFFT   8192
#define PI_F   3.14159265358979323846f

typedef __attribute__((ext_vector_type(16))) _Float16 v16h;
typedef __attribute__((ext_vector_type(8)))  float    v8f;

// ---------------------------------------------------------------------------
// WMMA fragment loader: 16-bit A/B matrix layout (CDNA5 ISA 7.12.2).
// For lanes 0-15:  halves 0..7 = K 0..7,  halves 8..15 = K 16..23
// For lanes 16-31: halves 0..7 = K 8..15, halves 8..15 = K 24..31
// Caller passes p = rowBase + k0 + ((lane>>4)*8); we read p[0..7], p[16..23].
// ---------------------------------------------------------------------------
__device__ __forceinline__ v16h frag_load_f32(const float* __restrict__ p) {
  const float4* q = (const float4*)p;
  float4 a = q[0], b = q[1], c = q[4], d = q[5];
  v16h f;
  f[0]  = (_Float16)a.x; f[1]  = (_Float16)a.y; f[2]  = (_Float16)a.z; f[3]  = (_Float16)a.w;
  f[4]  = (_Float16)b.x; f[5]  = (_Float16)b.y; f[6]  = (_Float16)b.z; f[7]  = (_Float16)b.w;
  f[8]  = (_Float16)c.x; f[9]  = (_Float16)c.y; f[10] = (_Float16)c.z; f[11] = (_Float16)c.w;
  f[12] = (_Float16)d.x; f[13] = (_Float16)d.y; f[14] = (_Float16)d.z; f[15] = (_Float16)d.w;
  return f;
}

// ---------------------------------------------------------------------------
// Kernel 1: implicit filter MLP -> k (D, L)
// ---------------------------------------------------------------------------
__global__ __launch_bounds__(256) void filter_kernel(
    const float* __restrict__ w1, const float* __restrict__ b1,
    const float* __restrict__ freq,
    const float* __restrict__ w2, const float* __restrict__ b2,
    const float* __restrict__ w3, const float* __restrict__ b3,
    const float* __restrict__ w4, float* __restrict__ k) {
  int l = blockIdx.x * 256 + threadIdx.x;
  if (l >= L_) return;
  float t = (float)l / (float)(L_ - 1);
  float w = 2.0f * PI_F * (float)l / (float)L_;
  // bands = 1, f = linspace(1e-4, 0, 1) = [1e-4]
  float z0 = t, z1 = __cosf(1e-4f * w), z2 = -__sinf(1e-4f * w);
  float h1[FORD_], h2[FORD_], h3[FORD_];
  for (int i = 0; i < FORD_; i++) {
    float a = w1[i*3+0]*z0 + w1[i*3+1]*z1 + w1[i*3+2]*z2 + b1[i];
    h1[i] = __sinf(freq[i] * a);
  }
  for (int i = 0; i < FORD_; i++) {
    float a = b2[i];
    for (int j = 0; j < FORD_; j++) a += w2[i*FORD_+j] * h1[j];
    h2[i] = __sinf(freq[i] * a);
  }
  for (int i = 0; i < FORD_; i++) {
    float a = b3[i];
    for (int j = 0; j < FORD_; j++) a += w3[i*FORD_+j] * h2[j];
    h3[i] = __sinf(freq[i] * a);
  }
  const float min_d = -4.6051701860f / 1.5f;   // log(0.01)/1.5
  const float max_d = -4.6051701860f / 0.3f;   // log(0.01)/0.3
  for (int d = 0; d < D_; d++) {
    float a = 0.0f;
    for (int f = 0; f < FORD_; f++) a += w4[d*FORD_+f] * h3[f];
    float delta = fabsf(min_d + (max_d - min_d) * (float)d / (float)(D_ - 1));
    k[(size_t)d * L_ + l] = a * __expf(-t * delta);
  }
}

// ---------------------------------------------------------------------------
// Mixed-radix LDS FFT, N = 8192 = 4^6 * 2, 256 threads.
// Forward: radix-4 DIF stages (len = 2048..2) + final radix-2 (digit-reversed
// output order). Inverse: exact stage mirror with conjugate twiddles
// (digit-reversed in -> natural out, unscaled; caller applies 1/N).
// Pointwise product is order-agnostic: k-hat uses the identical forward.
// ---------------------------------------------------------------------------
__device__ __forceinline__ void fft_fwd(float* re, float* im, int tid) {
  for (int len = NFFT / 4; len >= 2; len >>= 2) {
    for (int t = tid; t < NFFT / 4; t += 256) {
      int j    = t & (len - 1);
      int base = ((t & ~(len - 1)) << 2) | j;
      int i0 = base, i1 = base + len, i2 = base + 2 * len, i3 = base + 3 * len;
      float x0r = re[i0], x0i = im[i0];
      float x1r = re[i1], x1i = im[i1];
      float x2r = re[i2], x2i = im[i2];
      float x3r = re[i3], x3i = im[i3];
      float Ar = x0r + x2r, Ai = x0i + x2i;
      float Br = x0r - x2r, Bi = x0i - x2i;
      float Cr = x1r + x3r, Ci = x1i + x3i;
      float Dr = x1r - x3r, Di = x1i - x3i;
      re[i0] = Ar + Cr; im[i0] = Ai + Ci;                 // y0 = A + C
      float ang = -2.0f * PI_F * (float)j / (float)(4 * len);
      float s1, c1; __sincosf(ang, &s1, &c1);             // w1
      float c2 = c1 * c1 - s1 * s1, s2 = 2.0f * c1 * s1;  // w2 = w1^2
      float c3 = c2 * c1 - s2 * s1, s3 = c2 * s1 + s2 * c1; // w3 = w1*w2
      float t1r = Br + Di, t1i = Bi - Dr;                 // B - i*D
      re[i1] = t1r * c1 - t1i * s1; im[i1] = t1r * s1 + t1i * c1;
      float t2r = Ar - Cr, t2i = Ai - Ci;                 // A - C
      re[i2] = t2r * c2 - t2i * s2; im[i2] = t2r * s2 + t2i * c2;
      float t3r = Br - Di, t3i = Bi + Dr;                 // B + i*D
      re[i3] = t3r * c3 - t3i * s3; im[i3] = t3r * s3 + t3i * c3;
    }
    __syncthreads();
  }
  for (int t = tid; t < NFFT / 2; t += 256) {             // radix-2, len = 1
    int i0 = 2 * t, i1 = i0 + 1;
    float ar = re[i0], ai = im[i0], br = re[i1], bi = im[i1];
    re[i0] = ar + br; im[i0] = ai + bi;
    re[i1] = ar - br; im[i1] = ai - bi;
  }
  __syncthreads();
}

__device__ __forceinline__ void fft_inv(float* re, float* im, int tid) {
  for (int t = tid; t < NFFT / 2; t += 256) {             // radix-2, len = 1
    int i0 = 2 * t, i1 = i0 + 1;
    float ar = re[i0], ai = im[i0], br = re[i1], bi = im[i1];
    re[i0] = ar + br; im[i0] = ai + bi;
    re[i1] = ar - br; im[i1] = ai - bi;
  }
  __syncthreads();
  for (int len = 2; len <= NFFT / 4; len <<= 2) {
    for (int t = tid; t < NFFT / 4; t += 256) {
      int j    = t & (len - 1);
      int base = ((t & ~(len - 1)) << 2) | j;
      int i0 = base, i1 = base + len, i2 = base + 2 * len, i3 = base + 3 * len;
      float ang = 2.0f * PI_F * (float)j / (float)(4 * len); // conj twiddles
      float s1, c1; __sincosf(ang, &s1, &c1);
      float c2 = c1 * c1 - s1 * s1, s2 = 2.0f * c1 * s1;
      float c3 = c2 * c1 - s2 * s1, s3 = c2 * s1 + s2 * c1;
      float y0r = re[i0], y0i = im[i0];
      float y1r = re[i1] * c1 - im[i1] * s1, y1i = re[i1] * s1 + im[i1] * c1;
      float y2r = re[i2] * c2 - im[i2] * s2, y2i = re[i2] * s2 + im[i2] * c2;
      float y3r = re[i3] * c3 - im[i3] * s3, y3i = re[i3] * s3 + im[i3] * c3;
      float Ar = y0r + y2r, Ai = y0i + y2i;
      float Br = y0r - y2r, Bi = y0i - y2i;
      float Cr = y1r + y3r, Ci = y1i + y3i;
      float Dr = y1r - y3r, Di = y1i - y3i;
      re[i0] = Ar + Cr; im[i0] = Ai + Ci;                 // x0 = A + C
      re[i1] = Br - Di; im[i1] = Bi + Dr;                 // x1 = B + i*D
      re[i2] = Ar - Cr; im[i2] = Ai - Ci;                 // x2 = A - C
      re[i3] = Br + Di; im[i3] = Bi - Dr;                 // x3 = B - i*D
    }
    __syncthreads();
  }
}

// Kernel 2: k-hat = fwd-FFT(zero-pad(k[d])), stored digit-reversed, per d
__global__ __launch_bounds__(256) void khat_kernel(
    const float* __restrict__ k, float* __restrict__ kre, float* __restrict__ kim) {
  __shared__ float re[NFFT];
  __shared__ float im[NFFT];
  const int d = blockIdx.x, tid = threadIdx.x;
  for (int i = tid; i < NFFT; i += 256) {
    re[i] = (i < L_) ? k[(size_t)d * L_ + i] : 0.0f;
    im[i] = 0.0f;
  }
  __syncthreads();
  fft_fwd(re, im, tid);
  for (int i = tid; i < NFFT; i += 256) {
    kre[(size_t)d * NFFT + i] = re[i];
    kim[(size_t)d * NFFT + i] = im[i];
  }
}

// ---------------------------------------------------------------------------
// Kernel 3: x(B,L,INNER) = u(B,L,D) @ W_in^T + b_in   [WMMA f16->f32]
// WG = 256 thr = 8 waves arranged 4(M) x 2(N); macro tile 128(L) x 64(E); K=256.
// ---------------------------------------------------------------------------
__global__ __launch_bounds__(256) void gemm_in_kernel(
    const float* __restrict__ u, const float* __restrict__ W_in,
    const float* __restrict__ b_in, float* __restrict__ x) {
  const int tid  = threadIdx.x;
  const int lane = tid & 31;
  const int wave = tid >> 5;
  const int waveM = wave & 3, waveN = wave >> 2;
  const int b     = blockIdx.z;
  const int m_blk = blockIdx.x * 128 + waveM * 32;   // l
  const int n_blk = blockIdx.y * 64  + waveN * 32;   // e
  const int lo = lane & 15;
  const int kh = (lane >> 4) * 8;

  v8f c00 = {0,0,0,0,0,0,0,0}, c01 = c00, c10 = c00, c11 = c00;
  const float* ub = u + (size_t)b * L_ * D_;
  for (int k0 = 0; k0 < D_; k0 += 32) {
    const float* pa0 = ub + (size_t)(m_blk      + lo) * D_ + k0 + kh;
    const float* pa1 = ub + (size_t)(m_blk + 16 + lo) * D_ + k0 + kh;
    if (k0 + 32 < D_) {                           // prefetch next K tile
      __builtin_prefetch(pa0 + 32, 0, 3);
      __builtin_prefetch(pa1 + 32, 0, 3);
    }
    v16h a0 = frag_load_f32(pa0);
    v16h a1 = frag_load_f32(pa1);
    v16h b0 = frag_load_f32(W_in + (size_t)(n_blk      + lo) * D_ + k0 + kh);
    v16h b1 = frag_load_f32(W_in + (size_t)(n_blk + 16 + lo) * D_ + k0 + kh);
    c00 = __builtin_amdgcn_wmma_f32_16x16x32_f16(false, a0, false, b0, (short)0, c00, false, false);
    c01 = __builtin_amdgcn_wmma_f32_16x16x32_f16(false, a0, false, b1, (short)0, c01, false, false);
    c10 = __builtin_amdgcn_wmma_f32_16x16x32_f16(false, a1, false, b0, (short)0, c10, false, false);
    c11 = __builtin_amdgcn_wmma_f32_16x16x32_f16(false, a1, false, b1, (short)0, c11, false, false);
  }
  const int mh = (lane >> 4) * 8;
  float bias0 = b_in[n_blk + lo];
  float bias1 = b_in[n_blk + 16 + lo];
  float* xb = x + (size_t)b * L_ * INNER_;
#pragma unroll
  for (int r = 0; r < 8; r++) {
    int m0 = m_blk + r + mh;
    int m1 = m_blk + 16 + r + mh;
    xb[(size_t)m0 * INNER_ + n_blk      + lo] = c00[r] + bias0;
    xb[(size_t)m0 * INNER_ + n_blk + 16 + lo] = c01[r] + bias1;
    xb[(size_t)m1 * INNER_ + n_blk      + lo] = c10[r] + bias0;
    xb[(size_t)m1 * INNER_ + n_blk + 16 + lo] = c11[r] + bias1;
  }
}

// ---------------------------------------------------------------------------
// Kernel 4: depthwise causal 3-tap conv on x(B,L,INNER); emits x0 and v*x1,
// LDS-transposed to (B,D,L) so the FFT kernel reads channels contiguously.
// Grid: (L/64, D/32, B); 256 threads.
// ---------------------------------------------------------------------------
__global__ __launch_bounds__(256) void conv_split_kernel(
    const float* __restrict__ x, const float* __restrict__ short_w,
    const float* __restrict__ short_b,
    float* __restrict__ x0buf, float* __restrict__ vx1buf) {
  __shared__ float s0[32][65];
  __shared__ float s1[32][65];
  const int tid = threadIdx.x;
  const int b = blockIdx.z, d0 = blockIdx.y * 32, l0 = blockIdx.x * 64;
  const int dl = tid & 31;           // channel lane (coalesced along c)
  const int d  = d0 + dl;
  const int ls = tid >> 5;           // 8 l-slices per pass
  float w0[3], w1[3], w2[3];
#pragma unroll
  for (int j = 0; j < 3; j++) {
    w0[j] = short_w[(        d) * 3 + j];
    w1[j] = short_w[(D_   + d) * 3 + j];
    w2[j] = short_w[(2*D_ + d) * 3 + j];
  }
  const float sb0 = short_b[d], sb1 = short_b[D_ + d], sb2 = short_b[2*D_ + d];
  for (int li = ls; li < 64; li += 8) {
    int l = l0 + li;
    float a0 = 0.0f, a1 = 0.0f, a2 = 0.0f;
#pragma unroll
    for (int j = 0; j < 3; j++) {
      int lsrc = l - 2 + j;
      if (lsrc >= 0) {
        const float* row = x + (size_t)(b * L_ + lsrc) * INNER_;
        a0 += w0[j] * row[d];
        a1 += w1[j] * row[D_ + d];
        a2 += w2[j] * row[2*D_ + d];
      }
    }
    s0[dl][li] = a0 + sb0;                       // x0
    s1[dl][li] = (a1 + sb1) * (a2 + sb2);        // x1 * v
  }
  __syncthreads();
  // transposed, coalesced write-out: 32 rows x 64 l
  const int drow = tid >> 3;
  const int lch  = (tid & 7) * 8;
  float* p0 = x0buf  + ((size_t)(b * D_ + d0 + drow)) * L_ + l0 + lch;
  float* p1 = vx1buf + ((size_t)(b * D_ + d0 + drow)) * L_ + l0 + lch;
  float4 q0a = {s0[drow][lch+0], s0[drow][lch+1], s0[drow][lch+2], s0[drow][lch+3]};
  float4 q0b = {s0[drow][lch+4], s0[drow][lch+5], s0[drow][lch+6], s0[drow][lch+7]};
  float4 q1a = {s1[drow][lch+0], s1[drow][lch+1], s1[drow][lch+2], s1[drow][lch+3]};
  float4 q1b = {s1[drow][lch+4], s1[drow][lch+5], s1[drow][lch+6], s1[drow][lch+7]};
  ((float4*)p0)[0] = q0a; ((float4*)p0)[1] = q0b;
  ((float4*)p1)[0] = q1a; ((float4*)p1)[1] = q1b;
}

// ---------------------------------------------------------------------------
// Kernel 5: batch-pair packed FFT conv. For channel d and batches (2q, 2q+1):
//   z = vx1[2q] + i*vx1[2q+1];  w = ifft(fft(z) * khat[d]) = y0 + i*y1
// (both convs real, so real/imag parts separate exactly).  Then
//   y_b = (conv_b / N + vx1_b * fb) * x0_b.   One WG per (pair, d).
// ---------------------------------------------------------------------------
__global__ __launch_bounds__(256) void fftconv_kernel(
    const float* __restrict__ vx1, const float* __restrict__ x0buf,
    const float* __restrict__ kre, const float* __restrict__ kim,
    const float* __restrict__ filter_bias, float* __restrict__ ybuf) {
  __shared__ float re[NFFT];
  __shared__ float im[NFFT];
  const int d  = blockIdx.x & (D_ - 1);
  const int bp = blockIdx.x >> 8;          // batch pair 0..7
  const int b0 = 2 * bp, b1 = 2 * bp + 1;
  const int tid = threadIdx.x;
  const float* vch0 = vx1 + (size_t)(b0 * D_ + d) * L_;
  const float* vch1 = vx1 + (size_t)(b1 * D_ + d) * L_;
  for (int i = tid; i < NFFT; i += 256) {
    re[i] = (i < L_) ? vch0[i] : 0.0f;
    im[i] = (i < L_) ? vch1[i] : 0.0f;
  }
  __syncthreads();
  fft_fwd(re, im, tid);
  const float* krp = kre + (size_t)d * NFFT;
  const float* kip = kim + (size_t)d * NFFT;
  for (int i = tid; i < NFFT; i += 256) {
    float r = re[i], m = im[i];
    float wr = krp[i], wi = kip[i];
    re[i] = r * wr - m * wi;
    im[i] = r * wi + m * wr;
  }
  __syncthreads();
  fft_inv(re, im, tid);
  const float fb  = filter_bias[d];
  const float inv = 1.0f / (float)NFFT;
  const float* x0c0 = x0buf + (size_t)(b0 * D_ + d) * L_;
  const float* x0c1 = x0buf + (size_t)(b1 * D_ + d) * L_;
  float* yc0 = ybuf + (size_t)(b0 * D_ + d) * L_;
  float* yc1 = ybuf + (size_t)(b1 * D_ + d) * L_;
  for (int l = tid; l < L_; l += 256) {
    yc0[l] = (re[l] * inv + vch0[l] * fb) * x0c0[l];
    yc1[l] = (im[l] * inv + vch1[l] * fb) * x0c1[l];
  }
}

// ---------------------------------------------------------------------------
// Kernel 6: out(B,L,D) = y(B,D,L)^T @ W_out^T + b_out   [WMMA, LDS transpose]
// ---------------------------------------------------------------------------
__global__ __launch_bounds__(256) void gemm_out_kernel(
    const float* __restrict__ y, const float* __restrict__ W_out,
    const float* __restrict__ b_out, float* __restrict__ out) {
  __shared__ float ldsY[32][132];     // [k(d)][m(l)] tile, padded
  const int tid  = threadIdx.x;
  const int lane = tid & 31;
  const int wave = tid >> 5;
  const int waveM = wave & 3, waveN = wave >> 2;
  const int b     = blockIdx.z;
  const int l_blk = blockIdx.x * 128;
  const int m_loc = waveM * 32;
  const int n_blk = blockIdx.y * 64 + waveN * 32;
  const int lo = lane & 15;
  const int kh = (lane >> 4) * 8;

  v8f c00 = {0,0,0,0,0,0,0,0}, c01 = c00, c10 = c00, c11 = c00;
  for (int k0 = 0; k0 < D_; k0 += 32) {
    __syncthreads();
    {
      const int l_off = lane * 4;
#pragma unroll
      for (int it = 0; it < 4; it++) {
        int drow = (tid >> 5) + it * 8;
        float4 v = *(const float4*)(y + ((size_t)(b * D_ + k0 + drow)) * L_ + l_blk + l_off);
        ldsY[drow][l_off + 0] = v.x; ldsY[drow][l_off + 1] = v.y;
        ldsY[drow][l_off + 2] = v.z; ldsY[drow][l_off + 3] = v.w;
      }
    }
    __syncthreads();
    v16h a0, a1;
    {
      const int ml0 = m_loc + lo;
      const int ml1 = m_loc + 16 + lo;
#pragma unroll
      for (int i = 0; i < 8; i++) {
        a0[i]     = (_Float16)ldsY[kh + i][ml0];
        a0[i + 8] = (_Float16)ldsY[kh + 16 + i][ml0];
        a1[i]     = (_Float16)ldsY[kh + i][ml1];
        a1[i + 8] = (_Float16)ldsY[kh + 16 + i][ml1];
      }
    }
    v16h b0 = frag_load_f32(W_out + (size_t)(n_blk      + lo) * D_ + k0 + kh);
    v16h b1 = frag_load_f32(W_out + (size_t)(n_blk + 16 + lo) * D_ + k0 + kh);
    c00 = __builtin_amdgcn_wmma_f32_16x16x32_f16(false, a0, false, b0, (short)0, c00, false, false);
    c01 = __builtin_amdgcn_wmma_f32_16x16x32_f16(false, a0, false, b1, (short)0, c01, false, false);
    c10 = __builtin_amdgcn_wmma_f32_16x16x32_f16(false, a1, false, b0, (short)0, c10, false, false);
    c11 = __builtin_amdgcn_wmma_f32_16x16x32_f16(false, a1, false, b1, (short)0, c11, false, false);
  }
  const int mh = (lane >> 4) * 8;
  float bias0 = b_out[n_blk + lo];
  float bias1 = b_out[n_blk + 16 + lo];
  float* ob = out + (size_t)b * L_ * D_;
#pragma unroll
  for (int r = 0; r < 8; r++) {
    int l0v = l_blk + m_loc + r + mh;
    int l1v = l_blk + m_loc + 16 + r + mh;
    ob[(size_t)l0v * D_ + n_blk      + lo] = c00[r] + bias0;
    ob[(size_t)l0v * D_ + n_blk + 16 + lo] = c01[r] + bias1;
    ob[(size_t)l1v * D_ + n_blk      + lo] = c10[r] + bias0;
    ob[(size_t)l1v * D_ + n_blk + 16 + lo] = c11[r] + bias1;
  }
}

// ---------------------------------------------------------------------------
// Host-side launcher
// ---------------------------------------------------------------------------
extern "C" void kernel_launch(void* const* d_in, const int* in_sizes, int n_in,
                              void* d_out, int out_size, void* d_ws, size_t ws_size,
                              hipStream_t stream) {
  const float* u           = (const float*)d_in[0];
  const float* W_in        = (const float*)d_in[1];
  const float* b_in        = (const float*)d_in[2];
  const float* short_w     = (const float*)d_in[3];
  const float* short_b     = (const float*)d_in[4];
  const float* mlp_w1      = (const float*)d_in[5];
  const float* mlp_b1      = (const float*)d_in[6];
  const float* freq        = (const float*)d_in[7];
  const float* mlp_w2      = (const float*)d_in[8];
  const float* mlp_b2      = (const float*)d_in[9];
  const float* mlp_w3      = (const float*)d_in[10];
  const float* mlp_b3      = (const float*)d_in[11];
  const float* mlp_w4      = (const float*)d_in[12];
  const float* filter_bias = (const float*)d_in[13];
  const float* W_out       = (const float*)d_in[14];
  const float* b_out       = (const float*)d_in[15];
  float* out = (float*)d_out;

  // Workspace partition (bytes):
  //   x / y (reused):  B*L*INNER*4          = 201326592
  //   x0:              B*D*L*4              =  67108864
  //   v*x1:            B*D*L*4              =  67108864
  //   k:               D*L*4                =   4194304
  //   khat (re+im):    2*D*NFFT*4           =  16777216
  char* ws = (char*)d_ws;
  float* x      = (float*)(ws);                                   // also y
  float* x0buf  = (float*)(ws + 201326592ull);
  float* vx1buf = (float*)(ws + 201326592ull + 67108864ull);
  float* kbuf   = (float*)(ws + 201326592ull + 2ull * 67108864ull);
  float* kre    = (float*)(ws + 201326592ull + 2ull * 67108864ull + 4194304ull);
  float* kim    = kre + (size_t)D_ * NFFT;

  filter_kernel<<<L_ / 256, 256, 0, stream>>>(mlp_w1, mlp_b1, freq, mlp_w2, mlp_b2,
                                              mlp_w3, mlp_b3, mlp_w4, kbuf);
  khat_kernel<<<D_, 256, 0, stream>>>(kbuf, kre, kim);
  gemm_in_kernel<<<dim3(L_ / 128, INNER_ / 64, B_), 256, 0, stream>>>(u, W_in, b_in, x);
  conv_split_kernel<<<dim3(L_ / 64, D_ / 32, B_), 256, 0, stream>>>(x, short_w, short_b,
                                                                    x0buf, vx1buf);
  fftconv_kernel<<<(B_ / 2) * D_, 256, 0, stream>>>(vx1buf, x0buf, kre, kim,
                                                    filter_bias, x);
  gemm_out_kernel<<<dim3(L_ / 128, D_ / 64, B_), 256, 0, stream>>>(x, W_out, b_out, out);
}
